// VectorQuantizer_23021024707206
// MI455X (gfx1250) — compile-verified, matrix operand order checked by reading, not run on hardware
//
#include <hip/hip_runtime.h>

typedef __attribute__((ext_vector_type(16))) _Float16 v16h;
typedef __attribute__((ext_vector_type(8)))  _Float16 v8h;
typedef __attribute__((ext_vector_type(4)))  _Float16 v4h;
typedef __attribute__((ext_vector_type(8)))  float    v8f;

#define K_CODES 512
#define D_DIM   256
#define T_TOK   65536
#define ROW_H   264            // padded LDS row stride in halves (528 B -> 4-bank lane skew)
#define TOK_PER_PASS 64
#define PASSES  4

// LDS layout (bytes)
#define OFF_CB    0
#define SZ_CB     (K_CODES * ROW_H * 2)          // 270336
#define OFF_TOK   (OFF_CB + SZ_CB)
#define SZ_TOK    (TOK_PER_PASS * ROW_H * 2)     // 33792
#define OFF_WSQ   (OFF_TOK + SZ_TOK)
#define SZ_WSQ    (K_CODES * 4)
#define OFF_CV    (OFF_WSQ + SZ_WSQ)
#define SZ_CV     (TOK_PER_PASS * 8 * 4)
#define OFF_CI    (OFF_CV + SZ_CV)
#define SZ_CI     (TOK_PER_PASS * 8 * 4)
#define OFF_QS    (OFF_CI + SZ_CI)
#define SZ_QS     (TOK_PER_PASS * 4)
#define LDS_TOTAL (OFF_QS + SZ_QS)               // 310528 B < 320 KB

// ---------------- prep: codebook f32 -> f16, plus ||w||^2 ----------------
__global__ __launch_bounds__(256)
void vq_prep(const float* __restrict__ cb, _Float16* __restrict__ cb_h,
             float* __restrict__ wsq) {
  const int k = blockIdx.x;
  const int d = threadIdx.x;
  float v = cb[k * D_DIM + d];
  cb_h[k * D_DIM + d] = (_Float16)v;
  float s = v * v;
#pragma unroll
  for (int m = 16; m >= 1; m >>= 1) s += __shfl_xor(s, m, 32);
  __shared__ float part[8];
  if ((threadIdx.x & 31) == 0) part[threadIdx.x >> 5] = s;
  __syncthreads();
  if (threadIdx.x == 0) {
    float t = 0.f;
#pragma unroll
    for (int i = 0; i < 8; ++i) t += part[i];
    wsq[k] = t;
  }
}

// ---------------- main: distances via f16 WMMA, argmin, gather ----------------
__global__ __launch_bounds__(256)
void vq_main(const float* __restrict__ z_e, const float* __restrict__ cb,
             const _Float16* __restrict__ cb_h, const float* __restrict__ wsq,
             long long* __restrict__ qout, float* __restrict__ zq) {
  extern __shared__ __align__(16) char smem[];
  _Float16* cbL  = (_Float16*)(smem + OFF_CB);
  _Float16* tokL = (_Float16*)(smem + OFF_TOK);
  float*    wsqL = (float*)(smem + OFF_WSQ);
  float*    cvL  = (float*)(smem + OFF_CV);
  int*      ciL  = (int*)(smem + OFF_CI);
  int*      qsL  = (int*)(smem + OFF_QS);

  const int tid  = threadIdx.x;
  const int wave = tid >> 5;
  const int lane = tid & 31;
  const int r16  = lane & 15;
  const int half = lane >> 4;

  // ---- Stage f16 codebook into LDS via CDNA5 async copy (ASYNCcnt path) ----
  {
    // LDS byte offset of cbL within the workgroup's LDS allocation
    const unsigned cbL_base =
        (unsigned)(size_t)(__attribute__((address_space(3))) char*)(smem + OFF_CB);
    for (int c = tid; c < (K_CODES * D_DIM / 8); c += 256) {
      const int row = c >> 5, col = c & 31;
      const unsigned lds_b = cbL_base + (unsigned)(row * (ROW_H * 2) + col * 16);
      const _Float16* gp = cb_h + (row * D_DIM + col * 8);
      asm volatile("global_load_async_to_lds_b128 %0, %1, off"
                   :: "v"(lds_b), "v"(gp) : "memory");
    }
    asm volatile("s_wait_asynccnt 0x0" ::: "memory");
  }
  for (int k = tid; k < K_CODES; k += 256) wsqL[k] = wsq[k];
  __syncthreads();

  for (int pass = 0; pass < PASSES; ++pass) {
    const int tokBase = (blockIdx.x * PASSES + pass) * TOK_PER_PASS;
    const int img = tokBase >> 10;   // 1024 tokens per image
    const int p0  = tokBase & 1023;
    const float* zin = z_e + (size_t)img * (D_DIM * 1024) + p0;

    // Stage 64 tokens (f32 NCHW -> f16 LDS [tok][dim]); lanes span contiguous p,
    // 4 dims packed per ds_store_b64.
    {
      const int i = tid & 63, db = tid >> 6;
#pragma unroll 2
      for (int it = 0; it < 16; ++it) {
        const int d = it * 16 + db * 4;
        v4h p;
#pragma unroll
        for (int e = 0; e < 4; ++e) p[e] = (_Float16)zin[(d + e) * 1024 + i];
        *(v4h*)(tokL + i * ROW_H + d) = p;
      }
    }
    __syncthreads();

    // ---- GEMM: kc outer; B-tiles reused across 4 M-tiles, A across 4 N-tiles.
    // 16 live accumulators (128 VGPRs); exactly one ds_load_b128 per WMMA.
    v8f acc[4][4] = {};
#pragma unroll
    for (int kc = 0; kc < 8; ++kc) {
      v16h b[4];
#pragma unroll
      for (int j = 0; j < 4; ++j) {
        const _Float16* bp =
            cbL + (wave * 64 + j * 16 + r16) * ROW_H + kc * 32 + half * 8;
        const v8h blo = *(const v8h*)bp;
        const v8h bhi = *(const v8h*)(bp + 16);
#pragma unroll
        for (int e = 0; e < 8; ++e) { b[j][e] = blo[e]; b[j][e + 8] = bhi[e]; }
      }
#pragma unroll
      for (int mt = 0; mt < 4; ++mt) {
        const _Float16* ap =
            tokL + (mt * 16 + r16) * ROW_H + kc * 32 + half * 8;
        const v8h alo = *(const v8h*)ap;
        const v8h ahi = *(const v8h*)(ap + 16);
        v16h a;
#pragma unroll
        for (int e = 0; e < 8; ++e) { a[e] = alo[e]; a[e + 8] = ahi[e]; }
#pragma unroll
        for (int j = 0; j < 4; ++j)
          acc[mt][j] = __builtin_amdgcn_wmma_f32_16x16x32_f16(
              false, a, false, b[j], (short)0, acc[mt][j], false, false);
      }
    }

    // ---- argmin: dist = ||w||^2 - 2*dot (||x||^2 constant per row) ----
    float wq[4];
#pragma unroll
    for (int j = 0; j < 4; ++j) wq[j] = wsqL[wave * 64 + j * 16 + r16];

#pragma unroll
    for (int mt = 0; mt < 4; ++mt) {
#pragma unroll
      for (int r = 0; r < 8; ++r) {
        float bv = wq[0] - 2.0f * acc[mt][0][r];
        int   bi = wave * 64 + r16;
#pragma unroll
        for (int j = 1; j < 4; ++j) {
          const float v = wq[j] - 2.0f * acc[mt][j][r];
          if (v < bv) { bv = v; bi = wave * 64 + j * 16 + r16; }  // j asc => idx asc
        }
        // reduce across the 16 columns of this lane-half (index-carrying min)
#pragma unroll
        for (int off = 8; off >= 1; off >>= 1) {
          const float ov = __shfl_xor(bv, off, 32);
          const int   oi = __shfl_xor(bi, off, 32);
          if (ov < bv || (ov == bv && oi < bi)) { bv = ov; bi = oi; }
        }
        if (r16 == 0) {
          const int tokm = mt * 16 + half * 8 + r;  // C layout: VGPR r, halves = m / m+8
          cvL[tokm * 8 + wave] = bv;
          ciL[tokm * 8 + wave] = bi;
        }
      }
    }
    __syncthreads();

    // cross-wave argmin (waves own ascending code ranges -> tie-break by index)
    if (tid < TOK_PER_PASS) {
      float bv = cvL[tid * 8];
      int   bi = ciL[tid * 8];
#pragma unroll
      for (int w = 1; w < 8; ++w) {
        const float v = cvL[tid * 8 + w];
        const int   i2 = ciL[tid * 8 + w];
        if (v < bv || (v == bv && i2 < bi)) { bv = v; bi = i2; }
      }
      qout[tokBase + tid] = (long long)bi;
      qsL[tid] = bi;
    }
    __syncthreads();

    // gather z_q from f32 codebook, coalesced NCHW stores
    {
      const int i = tid & 63, db = tid >> 6;
      const int code = qsL[i];
      float* zo = zq + (size_t)img * (D_DIM * 1024) + p0;
#pragma unroll 4
      for (int d = db; d < D_DIM; d += 4)
        zo[d * 1024 + i] = cb[code * D_DIM + d];
    }
    __syncthreads();  // token tile reused next pass
  }
}

extern "C" void kernel_launch(void* const* d_in, const int* in_sizes, int n_in,
                              void* d_out, int out_size, void* d_ws, size_t ws_size,
                              hipStream_t stream) {
  const float* z_e = (const float*)d_in[0];
  const float* cb  = (const float*)d_in[1];

  _Float16* cb_h = (_Float16*)d_ws;                              // 256 KB
  float*    wsq  = (float*)((char*)d_ws + K_CODES * D_DIM * 2);  // +2 KB

  long long* qout = (long long*)d_out;                           // q: int64 [65536]
  float*     zq   = (float*)((char*)d_out + (size_t)T_TOK * 8);  // z_q: f32 NCHW

  vq_prep<<<K_CODES, 256, 0, stream>>>(cb, cb_h, wsq);
  vq_main<<<T_TOK / (TOK_PER_PASS * PASSES), 256, LDS_TOTAL, stream>>>(
      z_e, cb, cb_h, wsq, qout, zq);
}
